// Head_87196426043680
// MI455X (gfx1250) — matrix operand order repaired and checked by reading, not compiled, not run on hardware
//
#include <hip/hip_runtime.h>

typedef __bf16 bf16_t;
typedef __attribute__((ext_vector_type(16))) __bf16 v16bf;
typedef __attribute__((ext_vector_type(8)))  __bf16 v8bf;
typedef __attribute__((ext_vector_type(8)))  float  v8f;
typedef __attribute__((ext_vector_type(4)))  float  v4f;

#define T_SEQ 2048
#define D_MOD 1024
#define H_DIM 64
#define NBAT  8

// Load a 16-element bf16 fragment from two 16-byte chunks (VGPR0-3 / VGPR4-7).
static __device__ __forceinline__ v16bf ld_frag2(const bf16_t* p0, const bf16_t* p1) {
  v8bf lo = *(const v8bf*)p0;
  v8bf hi = *(const v8bf*)p1;
  v16bf r;
#pragma unroll
  for (int i = 0; i < 8; ++i) { r[i] = lo[i]; r[i + 8] = hi[i]; }
  return r;
}

static __device__ __forceinline__ v8f wmma_bf16(v16bf a, v16bf b, v8f c) {
  // D = A(16x32) * B(32x16) + C, fp32 accumulate
  return __builtin_amdgcn_wmma_f32_16x16x32_bf16(false, a, false, b, (short)0, c,
                                                 false, false);
}

// ---------------------------------------------------------------------------
// Kernel 0: W[1024][64] fp32 -> W^T[3][64][1024] bf16 (B-fragment friendly)
// ---------------------------------------------------------------------------
__global__ void wt_convert_kernel(const float* __restrict__ Wq,
                                  const float* __restrict__ Wk,
                                  const float* __restrict__ Wv,
                                  bf16_t* __restrict__ WT) {
  const int idx = blockIdx.x * blockDim.x + threadIdx.x;   // < 3*64*1024
  const int m = idx >> 16;
  const int r = idx & 65535;
  const int h = r >> 10;
  const int d = r & 1023;
  const float* W = (m == 0) ? Wq : ((m == 1) ? Wk : Wv);
  WT[idx] = (bf16_t)W[d * H_DIM + h];
}

// ---------------------------------------------------------------------------
// Kernel 1: QKV projection. One wave computes one 16x64 output tile of one of
// Q/K/V using v_wmma_f32_16x16x32_bf16. Q,K stored [B*T][64] bf16 row-major;
// V stored transposed per batch: Vt[b][64][2048] bf16.
// ---------------------------------------------------------------------------
__global__ void __launch_bounds__(256)
qkv_proj_kernel(const float* __restrict__ x, const bf16_t* __restrict__ WT,
                bf16_t* __restrict__ Q, bf16_t* __restrict__ Kc,
                bf16_t* __restrict__ Vt) {
  const int lane = threadIdx.x & 31;
  const int wid  = threadIdx.x >> 5;
  const int gw   = blockIdx.x * (blockDim.x >> 5) + wid;   // 0..3071
  const int mat  = gw >> 10;                               // 0=Q 1=K 2=V
  const int tile = gw & 1023;                              // 16-row tile
  const int row0 = tile << 4;
  const int half = lane >> 4;
  const int l16  = lane & 15;

  const bf16_t* W    = WT + (size_t)mat * (H_DIM * D_MOD);
  const float*  xrow = x + (size_t)(row0 + l16) * D_MOD;

  v8f acc[4];
#pragma unroll
  for (int n = 0; n < 4; ++n)
#pragma unroll
    for (int i = 0; i < 8; ++i) acc[n][i] = 0.0f;

  for (int k0 = 0; k0 < D_MOD; k0 += 32) {
    // A fragment: row = l16, K = {0..7,16..23} (half 0) / {8..15,24..31} (half 1)
    const int ka = k0 + half * 8;
    v4f xa0 = *(const v4f*)(xrow + ka);
    v4f xa1 = *(const v4f*)(xrow + ka + 4);
    v4f xb0 = *(const v4f*)(xrow + ka + 16);
    v4f xb1 = *(const v4f*)(xrow + ka + 20);
    v16bf a;
#pragma unroll
    for (int i = 0; i < 4; ++i) {
      a[i]      = (bf16_t)xa0[i];
      a[i + 4]  = (bf16_t)xa1[i];
      a[i + 8]  = (bf16_t)xb0[i];
      a[i + 12] = (bf16_t)xb1[i];
    }
    // B fragments: col = l16, K = {0..7,16..23} (half 0) / {8..15,24..31} (half 1), along d in W^T
    const int kb = k0 + half * 8;
#pragma unroll
    for (int n = 0; n < 4; ++n) {
      const bf16_t* wrow = W + (size_t)(n * 16 + l16) * D_MOD + kb;
      acc[n] = wmma_bf16(a, ld_frag2(wrow, wrow + 16), acc[n]);
    }
  }

  if (mat < 2) {
    bf16_t* out = (mat == 0) ? Q : Kc;
#pragma unroll
    for (int r = 0; r < 8; ++r) {
      const int row = row0 + r + half * 8;   // D-layout: row = reg + 8*(lane>=16)
#pragma unroll
      for (int n = 0; n < 4; ++n)
        out[(size_t)row * H_DIM + n * 16 + l16] = (bf16_t)acc[n][r];
    }
  } else {
    // Transposed V: per lane the 8 accumulator regs are 8 consecutive tokens
    // of one head column -> one contiguous b128 store per head-subtile.
    const int t0 = row0 + half * 8;
    const int b  = t0 >> 11;
    const int t  = t0 & (T_SEQ - 1);
#pragma unroll
    for (int n = 0; n < 4; ++n) {
      const int h = n * 16 + l16;
      v8bf pk;
#pragma unroll
      for (int r = 0; r < 8; ++r) pk[r] = (bf16_t)acc[n][r];
      *(v8bf*)(Vt + (size_t)b * H_DIM * T_SEQ + (size_t)h * T_SEQ + t) = pk;
    }
  }
}

// ---------------------------------------------------------------------------
// Kernel 2: flash attention. One wave owns a 16-query tile; loops over 32-key
// tiles up to the causal diagonal. Online softmax in fp32; score->probability
// tile crosses from C/D layout to A layout through a 1KB per-wave LDS buffer.
// ---------------------------------------------------------------------------
__global__ void __launch_bounds__(256)
attn_kernel(const bf16_t* __restrict__ Q, const bf16_t* __restrict__ Kc,
            const bf16_t* __restrict__ Vt, float* __restrict__ out) {
  __shared__ __attribute__((aligned(16))) bf16_t plds[8 * 16 * 32];
  const int lane = threadIdx.x & 31;
  const int wid  = threadIdx.x >> 5;
  const int gw   = blockIdx.x * 8 + wid;    // 0..1023
  const int b    = gw >> 7;
  const int qt   = gw & 127;
  const int q0   = qt << 4;
  const int half = lane >> 4;
  const int l16  = lane & 15;

  const bf16_t* Qb = Q  + (size_t)b * T_SEQ * H_DIM;
  const bf16_t* Kb = Kc + (size_t)b * T_SEQ * H_DIM;
  const bf16_t* Vb = Vt + (size_t)b * H_DIM * T_SEQ;

  // Q A-fragments for head dims 0-31 and 32-63 (persist across the k loop)
  const bf16_t* qrow = Qb + (size_t)(q0 + l16) * H_DIM + half * 16;
  const v16bf qa0 = ld_frag2(qrow, qrow + 8);
  const v16bf qa1 = ld_frag2(qrow + 32, qrow + 40);

  v8f o0, o1, o2, o3;
  float m_[8], l_[8];
#pragma unroll
  for (int r = 0; r < 8; ++r) {
    o0[r] = o1[r] = o2[r] = o3[r] = 0.0f;
    m_[r] = -3.0e38f;
    l_[r] = 0.0f;
  }

  bf16_t* myP = plds + wid * (16 * 32);
  const float scale  = 0.125f;      // 1/sqrt(64)
  const float NEGINF = -3.0e38f;

  for (int kt = 0; kt < q0 + 16; kt += 32) {
    // ---- S = Q * K^T  (16 queries x 32 keys), 4 WMMAs ----
    v8f s0, s1;
#pragma unroll
    for (int r = 0; r < 8; ++r) { s0[r] = 0.0f; s1[r] = 0.0f; }
    {
      const bf16_t* k0p = Kb + (size_t)(kt + l16) * H_DIM + half * 16;
      s0 = wmma_bf16(qa0, ld_frag2(k0p, k0p + 8), s0);
      s0 = wmma_bf16(qa1, ld_frag2(k0p + 32, k0p + 40), s0);
      const bf16_t* k1p = Kb + (size_t)(kt + 16 + l16) * H_DIM + half * 16;
      s1 = wmma_bf16(qa0, ld_frag2(k1p, k1p + 8), s1);
      s1 = wmma_bf16(qa1, ld_frag2(k1p + 32, k1p + 40), s1);
    }

    // ---- online softmax (rows live across the 16 lanes of each half) ----
    const int key0 = kt + l16;
    const int key1 = kt + 16 + l16;
    float p0[8], p1[8], corr[8];
#pragma unroll
    for (int r = 0; r < 8; ++r) {
      const int qr = q0 + r + half * 8;
      float a = s0[r] * scale;
      float c = s1[r] * scale;
      if (key0 > qr) a = NEGINF;           // causal mask
      if (key1 > qr) c = NEGINF;
      float mx = fmaxf(a, c);
#pragma unroll
      for (int off = 1; off < 16; off <<= 1)
        mx = fmaxf(mx, __shfl_xor(mx, off, 32));
      const float mnew = fmaxf(m_[r], mx);
      const float pa = __expf(a - mnew);
      const float pb = __expf(c - mnew);
      float rs = pa + pb;
#pragma unroll
      for (int off = 1; off < 16; off <<= 1)
        rs += __shfl_xor(rs, off, 32);
      const float cr = __expf(m_[r] - mnew);
      l_[r] = l_[r] * cr + rs;
      m_[r] = mnew;
      corr[r] = cr;
      p0[r] = pa;
      p1[r] = pb;
    }
#pragma unroll
    for (int r = 0; r < 8; ++r) {
      o0[r] *= corr[r]; o1[r] *= corr[r]; o2[r] *= corr[r]; o3[r] *= corr[r];
    }

    // ---- P: C/D layout -> A layout through per-wave LDS tile ----
#pragma unroll
    for (int r = 0; r < 8; ++r) {
      myP[(r + half * 8) * 32 + l16]      = (bf16_t)p0[r];
      myP[(r + half * 8) * 32 + 16 + l16] = (bf16_t)p1[r];
    }
    // DS ops are in-order per wave; wait for the stores, block reordering.
    asm volatile("s_wait_dscnt 0" ::: "memory");
    const bf16_t* prow = myP + l16 * 32 + half * 16;
    const v16bf pfrag = ld_frag2(prow, prow + 8);

    // ---- O += P * V (V pre-transposed -> contiguous B-fragment loads) ----
    {
      const bf16_t* v0p = Vb + (size_t)l16 * T_SEQ + kt + half * 16;
      o0 = wmma_bf16(pfrag, ld_frag2(v0p, v0p + 8), o0);
      const bf16_t* v1p = v0p + (size_t)16 * T_SEQ;
      o1 = wmma_bf16(pfrag, ld_frag2(v1p, v1p + 8), o1);
      const bf16_t* v2p = v0p + (size_t)32 * T_SEQ;
      o2 = wmma_bf16(pfrag, ld_frag2(v2p, v2p + 8), o2);
      const bf16_t* v3p = v0p + (size_t)48 * T_SEQ;
      o3 = wmma_bf16(pfrag, ld_frag2(v3p, v3p + 8), o3);
    }
  }

  // ---- finalize: divide by row sums, store fp32 output ----
  float* ob = out + (size_t)b * T_SEQ * H_DIM;
#pragma unroll
  for (int r = 0; r < 8; ++r) {
    const int row = q0 + r + half * 8;
    const float inv = 1.0f / l_[r];
    ob[(size_t)row * H_DIM + l16]      = o0[r] * inv;
    ob[(size_t)row * H_DIM + 16 + l16] = o1[r] * inv;
    ob[(size_t)row * H_DIM + 32 + l16] = o2[r] * inv;
    ob[(size_t)row * H_DIM + 48 + l16] = o3[r] * inv;
  }
}

// ---------------------------------------------------------------------------
extern "C" void kernel_launch(void* const* d_in, const int* in_sizes, int n_in,
                              void* d_out, int out_size, void* d_ws,
                              size_t ws_size, hipStream_t stream) {
  const float* x  = (const float*)d_in[0];
  const float* Wq = (const float*)d_in[1];
  const float* Wk = (const float*)d_in[2];
  const float* Wv = (const float*)d_in[3];
  float* out = (float*)d_out;

  char* ws = (char*)d_ws;
  bf16_t* WT = (bf16_t*)(ws);                       // 3*64*1024*2  = 384 KB
  bf16_t* Qb = (bf16_t*)(ws + (size_t)0x080000);    // 16384*64*2  = 2 MB
  bf16_t* Kb = (bf16_t*)(ws + (size_t)0x280000);    // 2 MB
  bf16_t* Vt = (bf16_t*)(ws + (size_t)0x480000);    // 2 MB (transposed V)

  // 1) weight transpose + bf16 convert: 3*64*1024 elements
  wt_convert_kernel<<<768, 256, 0, stream>>>(Wq, Wk, Wv, WT);

  // 2) QKV projection: 3 matrices * 1024 row-tiles = 3072 waves, 8 waves/block
  qkv_proj_kernel<<<384, 256, 0, stream>>>(x, WT, Qb, Kb, Vt);

  // 3) flash attention: 8 batches * 128 query tiles = 1024 waves, 8 waves/block
  attn_kernel<<<128, 256, 0, stream>>>(Qb, Kb, Vt, out);
}
